// Classifier_5712306504361
// MI455X (gfx1250) — matrix operand order repaired and checked by reading, not compiled, not run on hardware
//
#include <hip/hip_runtime.h>
#include <stdint.h>

typedef __attribute__((ext_vector_type(16))) _Float16 v16h;
typedef __attribute__((ext_vector_type(8)))  float    v8f;
typedef __attribute__((ext_vector_type(4)))  float    v4f;

#define EPSR 1e-7f

__device__ __forceinline__ float arcosh_ref(float x) {
    x = fmaxf(x, 1.0f + EPSR);
    return logf(x + sqrtf(x * x - 1.0f));
}

template <int BD>
__device__ __forceinline__ float block_reduce_sum(float v, float* red) {
    #pragma unroll
    for (int o = 16; o > 0; o >>= 1) v += __shfl_xor(v, o, 32);
    const int wid = threadIdx.x >> 5, lid = threadIdx.x & 31;
    if (lid == 0) red[wid] = v;
    __syncthreads();
    float s = 0.0f;
    if (threadIdx.x == 0) {
        #pragma unroll
        for (int i = 0; i < BD / 32; ++i) s += red[i];
        red[0] = s;
    }
    __syncthreads();
    s = red[0];
    __syncthreads();
    return s;
}

// ---------------- utility kernels ----------------

__global__ void zero_f32_kernel(float* __restrict__ p, int n) {
    int i = blockIdx.x * blockDim.x + threadIdx.x;
    if (i < n) p[i] = 0.0f;
}

// Pad+convert W (OUT_TAIL x IN_TAIL, f32, row-major) into Wt (N_PAD x K_PAD, f16)
__global__ void prep_w_kernel(const float* __restrict__ W, _Float16* __restrict__ Wt,
                              int OUT_TAIL, int IN_TAIL, int K_PAD, int total) {
    int i = blockIdx.x * blockDim.x + threadIdx.x;
    if (i >= total) return;
    int n = i / K_PAD, k = i - n * K_PAD;
    float v = (n < OUT_TAIL && k < IN_TAIL) ? W[n * IN_TAIL + k] : 0.0f;
    Wt[i] = (_Float16)v;
}

// xt = log_map_zero(x); one wave per row of 128
__global__ __launch_bounds__(256) void logmap_x_kernel(const float* __restrict__ x,
                                                       float* __restrict__ xt, int N) {
    int row  = blockIdx.x * 8 + (threadIdx.x >> 5);
    int lane = threadIdx.x & 31;
    if (row >= N) return;
    const float* xr = x + (size_t)row * 128;
    float p = 0.0f;
    #pragma unroll
    for (int i = 0; i < 4; ++i) {
        int j = 1 + lane + 32 * i;
        if (j < 128) { float t = xr[j]; p += t * t; }
    }
    #pragma unroll
    for (int o = 16; o > 0; o >>= 1) p += __shfl_xor(p, o, 32);
    float nrm  = sqrtf(p + EPSR);
    float dist = arcosh_ref(xr[0] + EPSR);   // sc = 1
    float g    = dist / nrm;
    float* orow = xt + (size_t)row * 128;
    if (lane == 0) orow[0] = 0.0f;
    #pragma unroll
    for (int i = 0; i < 4; ++i) {
        int j = 1 + lane + 32 * i;
        if (j < 128) orow[j] = g * xr[j];
    }
}

// segment_sum: agg[dst] += xt[src]; 32 lanes per edge
__global__ __launch_bounds__(256) void aggregate_kernel(const int* __restrict__ ei,
                                                        const float* __restrict__ xt,
                                                        float* __restrict__ agg, int E) {
    int idx  = blockIdx.x * blockDim.x + threadIdx.x;
    int edge = idx >> 5, lane = idx & 31;
    if (edge >= E) return;
    int src = ei[edge], dst = ei[E + edge];
    const float* s = xt + (size_t)src * 128;
    float*       d = agg + (size_t)dst * 128;
    #pragma unroll
    for (int i = 0; i < 4; ++i) atomicAdd(&d[lane + 32 * i], s[lane + 32 * i]);
}

// out = xt + agg (tail), then exp_map_zero then log_map_zero -> f16 A-matrix (N x 128)
__global__ __launch_bounds__(256) void combine_explog_kernel(const float* __restrict__ xt,
                                                             const float* __restrict__ agg,
                                                             _Float16* __restrict__ Xa, int N) {
    int row  = blockIdx.x * 8 + (threadIdx.x >> 5);
    int lane = threadIdx.x & 31;
    if (row >= N) return;
    const float* xr = xt + (size_t)row * 128;
    const float* ar = agg + (size_t)row * 128;
    float t[4]; float p = 0.0f;
    #pragma unroll
    for (int i = 0; i < 4; ++i) {
        int j = 1 + lane + 32 * i;
        float v = (j < 128) ? (xr[j] + ar[j]) : 0.0f;
        t[i] = v; p += v * v;
    }
    #pragma unroll
    for (int o = 16; o > 0; o >>= 1) p += __shfl_xor(p, o, 32);
    float n1    = sqrtf(fmaxf(p + EPSR, 1e-6f));
    float f1    = sinhf(fminf(n1, 50.0f)) / n1;
    float sh2   = f1 * f1 * p;
    float first = sqrtf(1.0f + sh2);
    float dist  = arcosh_ref(first + EPSR);
    float scale = (dist / sqrtf(sh2 + EPSR)) * f1;
    _Float16* orow = Xa + (size_t)row * 128;
    #pragma unroll
    for (int i = 0; i < 4; ++i) {
        int j = 1 + lane + 32 * i;
        if (j < 128) orow[j - 1] = (_Float16)(scale * t[i]);
    }
    if (lane == 0) orow[127] = (_Float16)0.0f;
}

// ---------------- fused WMMA layer: t = A @ Wt^T + b, then Lorentz nonlinearity chain ----------------
// Block: 16 rows x N_PAD cols. A tile staged to LDS once per block via async copy
// (global_load_async_to_lds_b128 + s_wait_asynccnt); one wave per 16x16 WMMA tile.
template <int K_PAD, int N_PAD, int N_TAIL, bool LAST>
__global__ __launch_bounds__(N_PAD * 2) void lorentz_layer_kernel(
    const _Float16* __restrict__ A, const _Float16* __restrict__ Wt,
    const float* __restrict__ b, _Float16* __restrict__ Xnext,
    float* __restrict__ hmsum) {
    constexpr int BD = N_PAD * 2;
    __shared__ __attribute__((aligned(16))) _Float16 aTile[16 * K_PAD];
    __shared__ float tTile[16 * N_PAD];
    __shared__ float colsum[LAST ? N_PAD : 1];
    __shared__ float red[BD / 32];

    const int tid  = threadIdx.x;
    const int wave = tid >> 5, lane = tid & 31;
    const int rowBase = blockIdx.x * 16;
    const int colBase = wave * 16;
    const int mrow = lane & 15, half = lane >> 4;

    // ---- stage A tile (16 x K_PAD f16) into LDS with gfx1250 async copy ----
    constexpr int NCHUNK = (16 * K_PAD * 2) / 16;   // 16B chunks; NCHUNK <= BD for all shapes
    if (tid < NCHUNK) {
        const char* gsrc = (const char*)(A + (size_t)rowBase * K_PAD) + tid * 16;
        uint32_t ldst = (uint32_t)(uintptr_t)(&aTile[0]) + tid * 16;  // flat->LDS: low 32 bits
        asm volatile("global_load_async_to_lds_b128 %0, %1, off"
                     :: "v"(ldst), "v"((uint64_t)(uintptr_t)gsrc)
                     : "memory");
    }
    asm volatile("s_wait_asynccnt 0x0" ::: "memory");
    __syncthreads();

    // ---- GEMM (wave-level 16x16 tile), A from LDS, B (weights) from L2-resident global ----
    v8f acc = {};
    const _Float16* aRow = aTile + mrow * K_PAD;
    const _Float16* bRow = Wt + (size_t)(colBase + mrow) * K_PAD;  // B[k][n] = Wt[n][k]
    #pragma unroll 4
    for (int kt = 0; kt < K_PAD; kt += 32) {
        v16h afrag, bfrag;
        // A 16x32 f16 layout: per lane two 8-element chunks at k = 8*half and k = 8*half + 16
        const int ka = kt + 8 * half;
        ((v4f*)&afrag)[0] = *(const v4f*)(aRow + ka);
        ((v4f*)&afrag)[1] = *(const v4f*)(aRow + ka + 16);
        // B 32x16 f16 layout: per lane two 8-element chunks at k = 8*half and k = 8*half + 16
        const int kb = kt + 8 * half;
        ((v4f*)&bfrag)[0] = *(const v4f*)(bRow + kb);
        ((v4f*)&bfrag)[1] = *(const v4f*)(bRow + kb + 16);
        acc = __builtin_amdgcn_wmma_f32_16x16x32_f16(false, afrag, false, bfrag,
                                                     (short)0, acc, false, false);
    }
    // D layout: lane gives col = lane&15; VGPR r -> row r + 8*half. Add bias, stage in LDS.
    {
        const int n = colBase + mrow;
        const float bias = (n < N_TAIL) ? b[n] : 0.0f;
        const int mOff = 8 * half;
        #pragma unroll
        for (int r = 0; r < 8; ++r)
            tTile[(mOff + r) * N_PAD + n] = acc[r] + bias;
    }
    if (LAST) {
        for (int j = tid; j < N_PAD; j += BD) colsum[j] = 0.0f;
    }
    __syncthreads();

    // ---- epilogue: exp_map_zero -> log_map_zero -> relu -> exp_map_zero -> log_map_zero ----
    for (int r = 0; r < 16; ++r) {
        float* trow = &tTile[r * N_PAD];
        float p = 0.0f;
        for (int j = tid; j < N_PAD; j += BD) { float t = trow[j]; p += t * t; }
        const float sum_t2 = block_reduce_sum<BD>(p, red);
        const float n1 = sqrtf(fmaxf(sum_t2 + EPSR, 1e-6f));
        const float f1 = sinhf(fminf(n1, 50.0f)) / n1;
        const float sh2 = f1 * f1 * sum_t2;                 // ||tail_out||^2
        const float dist = arcosh_ref(sqrtf(1.0f + sh2) + EPSR);
        const float g1 = dist / sqrtf(sh2 + EPSR);
        float pu = 0.0f;
        for (int j = tid; j < N_PAD; j += BD) {
            float u = fmaxf(g1 * f1 * trow[j], 0.0f);       // relu(log_map tail)
            trow[j] = u;
            pu += u * u;
        }
        const float sum_u2 = block_reduce_sum<BD>(pu, red);
        const float n2 = sqrtf(fmaxf(sum_u2 + EPSR, 1e-6f));
        const float f2 = sinhf(fminf(n2, 50.0f)) / n2;
        const float sh2b = f2 * f2 * sum_u2;
        const float dist2 = arcosh_ref(sqrtf(1.0f + sh2b) + EPSR);
        const float scale = (dist2 / sqrtf(sh2b + EPSR)) * f2;  // xt_next = scale * u
        if (LAST) {
            for (int j = tid; j < N_PAD; j += BD) colsum[j] += scale * trow[j];
        } else {
            _Float16* orow = Xnext + (size_t)(rowBase + r) * N_PAD;
            for (int j = tid; j < N_PAD; j += BD) orow[j] = (_Float16)(scale * trow[j]);
        }
        __syncthreads();
    }
    if (LAST) {
        for (int j = tid; j < N_TAIL; j += BD) atomicAdd(&hmsum[1 + j], colsum[j]);
    }
}

// ---------------- classifier: mean -> lorentz_linear(Wc) -> lorentz_act(softmax) ----------------
__global__ __launch_bounds__(256) void classifier_kernel(const float* __restrict__ hmsum,
                                                         const float* __restrict__ Wc,
                                                         const float* __restrict__ bc,
                                                         float* __restrict__ out, int N) {
    __shared__ float xs[383];
    __shared__ float red[8];
    __shared__ float mx[9];
    const int tid = threadIdx.x;
    const float invN = 1.0f / (float)N;
    float p = 0.0f;
    for (int j = tid; j < 383; j += 256) {
        float v = hmsum[1 + j] * invN;   // hm tail; hm[0] = 0
        xs[j] = v; p += v * v;
    }
    const float s = block_reduce_sum<256>(p, red);
    const float g = arcosh_ref(0.0f + EPSR) / sqrtf(s + EPSR);  // dist/nrm of log_map_zero(hm)
    for (int o = 0; o < 9; ++o) {
        float q = 0.0f;
        for (int j = tid; j < 383; j += 256) q += g * xs[j] * Wc[o * 383 + j];
        float m = block_reduce_sum<256>(q, red);
        if (tid == 0) mx[o] = m + bc[o];
    }
    __syncthreads();
    if (tid == 0) {
        float sum2 = 0.0f; bool allz = true;
        for (int o = 0; o < 9; ++o) { sum2 += mx[o] * mx[o]; if (mx[o] != 0.0f) allz = false; }
        float hc[10];
        if (allz) { for (int i = 0; i < 10; ++i) hc[i] = 0.0f; }
        else {
            float n1 = sqrtf(fmaxf(sum2 + EPSR, 1e-6f));
            float f1 = sinhf(fminf(n1, 50.0f)) / n1;
            hc[0] = sqrtf(1.0f + f1 * f1 * sum2);
            for (int o = 0; o < 9; ++o) hc[1 + o] = f1 * mx[o];
        }
        for (int i = 0; i < 10; ++i) out[i] = hc[i];
        // lorentz_act with softmax
        float tail2 = 0.0f;
        for (int o = 1; o < 10; ++o) tail2 += hc[o] * hc[o];
        float gg = arcosh_ref(hc[0] + EPSR) / sqrtf(tail2 + EPSR);
        float lg[10]; lg[0] = 0.0f;
        for (int o = 1; o < 10; ++o) lg[o] = gg * hc[o];
        float mmax = lg[0];
        for (int i = 1; i < 10; ++i) mmax = fmaxf(mmax, lg[i]);
        float e[10], es = 0.0f;
        for (int i = 0; i < 10; ++i) { e[i] = expf(lg[i] - mmax); es += e[i]; }
        float u2 = 0.0f, u[10]; u[0] = 0.0f;
        for (int i = 1; i < 10; ++i) { u[i] = e[i] / es; u2 += u[i] * u[i]; }
        float n3 = sqrtf(fmaxf(u2 + EPSR, 1e-6f));
        float f3 = sinhf(fminf(n3, 50.0f)) / n3;
        out[10] = sqrtf(1.0f + f3 * f3 * u2);
        for (int i = 1; i < 10; ++i) out[10 + i] = f3 * u[i];
    }
}

extern "C" void kernel_launch(void* const* d_in, const int* in_sizes, int n_in,
                              void* d_out, int out_size, void* d_ws, size_t ws_size,
                              hipStream_t stream) {
    const float* x  = (const float*)d_in[0];
    const int*   ei = (const int*)d_in[1];
    const float* W0 = (const float*)d_in[2]; const float* b0 = (const float*)d_in[3];
    const float* W1 = (const float*)d_in[4]; const float* b1 = (const float*)d_in[5];
    const float* W2 = (const float*)d_in[6]; const float* b2 = (const float*)d_in[7];
    const float* Wc = (const float*)d_in[8]; const float* bc = (const float*)d_in[9];

    const int N = in_sizes[0] / 128;   // 50000
    const int E = in_sizes[1] / 2;     // 800000

    char* ws = (char*)d_ws;
    size_t ofs = 0;
    auto alloc = [&](size_t bytes) { void* p = ws + ofs; ofs = (ofs + bytes + 255) & ~(size_t)255; return p; };
    float*     xt    = (float*)alloc((size_t)N * 128 * 4);
    float*     agg   = (float*)alloc((size_t)N * 128 * 4);
    _Float16*  bufA  = (_Float16*)alloc((size_t)N * 256 * 2);
    _Float16*  bufB  = (_Float16*)alloc((size_t)N * 256 * 2);
    _Float16*  Wt0   = (_Float16*)alloc((size_t)128 * 128 * 2);
    _Float16*  Wt1   = (_Float16*)alloc((size_t)256 * 128 * 2);
    _Float16*  Wt2   = (_Float16*)alloc((size_t)384 * 256 * 2);
    float*     hmsum = (float*)alloc(384 * 4);

    const int nAgg = N * 128;
    zero_f32_kernel<<<(nAgg + 255) / 256, 256, 0, stream>>>(agg, nAgg);
    zero_f32_kernel<<<2, 256, 0, stream>>>(hmsum, 384);

    prep_w_kernel<<<(128 * 128 + 255) / 256, 256, 0, stream>>>(W0, Wt0, 127, 127, 128, 128 * 128);
    prep_w_kernel<<<(256 * 128 + 255) / 256, 256, 0, stream>>>(W1, Wt1, 255, 127, 128, 256 * 128);
    prep_w_kernel<<<(384 * 256 + 255) / 256, 256, 0, stream>>>(W2, Wt2, 383, 255, 256, 384 * 256);

    logmap_x_kernel<<<(N + 7) / 8, 256, 0, stream>>>(x, xt, N);
    aggregate_kernel<<<(E * 32 + 255) / 256, 256, 0, stream>>>(ei, xt, agg, E);
    combine_explog_kernel<<<(N + 7) / 8, 256, 0, stream>>>(xt, agg, bufA, N);

    const int gRows = N / 16;  // 3125, exact
    lorentz_layer_kernel<128, 128, 127, false><<<gRows, 256, 0, stream>>>(bufA, Wt0, b0, bufB, nullptr);
    lorentz_layer_kernel<128, 256, 255, false><<<gRows, 512, 0, stream>>>(bufB, Wt1, b1, bufA, nullptr);
    lorentz_layer_kernel<256, 384, 383, true ><<<gRows, 768, 0, stream>>>(bufA, Wt2, b2, nullptr, hmsum);

    classifier_kernel<<<1, 256, 0, stream>>>(hmsum, Wc, bc, (float*)d_out, N);
}
